// Attn_10041633538233
// MI455X (gfx1250) — compile-verified, matrix operand order checked by reading, not run on hardware
//
#include <hip/hip_runtime.h>
#include <hip/hip_bf16.h>

typedef __attribute__((ext_vector_type(2)))  float  v2f;
typedef __attribute__((ext_vector_type(4)))  float  v4f;
typedef __attribute__((ext_vector_type(8)))  float  v8f;
typedef __attribute__((ext_vector_type(16))) __bf16 v16bf;

static constexpr int Bsz  = 4;
static constexpr int C    = 256;
static constexpr int CQ   = 64;
static constexpr int N    = 4096;   // 64*64
static constexpr int LDSW = 4100;   // row stride in LDS tile (bank-conflict pad)

// ---------------------------------------------------------------------------
// K1: q/k projections (1x1 conv == per-pixel linear), stored TRANSPOSED as
// qT/kT [b][n][c] so the energy kernel's WMMA fragments are per-lane
// contiguous (b64 loads). Lane-varying n -> coalesced x loads; o is
// block-uniform so w_q/w_k become scalar loads.
// ---------------------------------------------------------------------------
__global__ __launch_bounds__(256) void proj_qk(
    const float* __restrict__ x,
    const float* __restrict__ wq, const float* __restrict__ bq,
    const float* __restrict__ wk, const float* __restrict__ bk,
    float* __restrict__ qT, float* __restrict__ kT)
{
  int tid   = threadIdx.x;
  int blk   = blockIdx.x;        // 4096 blocks = (B*CQ) * (N/256)
  int chunk = blk & 15;          // 16 chunks of 256 along N
  int bo    = blk >> 4;          // b*64 + o
  int o     = bo & 63;
  int b     = bo >> 6;
  int n     = chunk * 256 + tid;

  const float* xb  = x + (size_t)b * C * N + n;
  const float* wqo = wq + o * C;
  const float* wko = wk + o * C;
  float aq = bq[o];
  float ak = bk[o];
#pragma unroll 8
  for (int c = 0; c < C; ++c) {
    float xv = xb[(size_t)c * N];
    aq = fmaf(wqo[c], xv, aq);
    ak = fmaf(wko[c], xv, ak);
  }
  size_t t = ((size_t)b * N + n) * CQ + o;   // [b][n][c] layout
  qT[t] = aq;
  kT[t] = ak;
}

// ---------------------------------------------------------------------------
// K2 (fused): energy + softmax.
// One 8-wave block owns a full 16(i) x 4096(j) strip of energy[b]:
//   - each wave computes its 512-column slice with V_WMMA_F32_16X16X4_F32
//     (A panel of 16x64 f32 loaded once per wave, 16 WMMAs per 16x16 tile),
//     accumulators parked in a 16 x LDSW f32 LDS tile (256 KB < 320 KB WGP LDS).
//   - after the block barrier each wave owns 2 complete rows -> softmax is
//     wave-local (shuffle reductions), normalized rows stored b128-coalesced.
// This writes attention to HBM exactly once (no energy round-trip).
// f32 16x4 A layout: lane r=lane%16 holds row M=r; vgpr v holds K=v+2g.
// C/D layout: vgpr p -> M = p + 8g, N = lane%16.
// ---------------------------------------------------------------------------
__global__ __launch_bounds__(256) void energy_softmax(
    const float* __restrict__ qT, const float* __restrict__ kT,
    float* __restrict__ att)
{
  extern __shared__ float sm[];            // [16][LDSW]
  const int lane = threadIdx.x & 31;
  const int wave = threadIdx.x >> 5;
  const int g = lane >> 4;
  const int r = lane & 15;

  int blk = blockIdx.x;                    // 1024 blocks = B * 256 i-tiles
  int b   = blk >> 8;
  int i0  = (blk & 255) * 16;

  const float* qrow = qT + ((size_t)b * N + i0 + r) * CQ;
  v2f a[16];
#pragma unroll
  for (int s = 0; s < 16; ++s)
    a[s] = *(const v2f*)(qrow + 4 * s + 2 * g);          // K = 4s+2g, +1

  // ---- energy: 32 j-tiles per wave, accumulate into LDS tile ----
  for (int jt = 0; jt < 32; ++jt) {
    int j0 = wave * 512 + jt * 16;
    const float* krow = kT + ((size_t)b * N + j0 + r) * CQ;
    v8f acc = {};
#pragma unroll
    for (int s = 0; s < 16; ++s) {
      v2f bv = *(const v2f*)(krow + 4 * s + 2 * g);
      acc = __builtin_amdgcn_wmma_f32_16x16x4_f32(
          false, a[s], false, bv, (short)0, acc, false, false);
    }
#pragma unroll
    for (int p = 0; p < 8; ++p)
      sm[(p + 8 * g) * LDSW + j0 + r] = acc[p];
  }
  __syncthreads();

  // ---- softmax: wave w owns rows 2w, 2w+1 (wave-local reductions) ----
  float* ab = att + (size_t)b * N * N;
  for (int rr = 0; rr < 2; ++rr) {
    int row = wave * 2 + rr;
    float* srow = sm + row * LDSW;

    float m = -3.402823466e38f;
#pragma unroll 4
    for (int t = 0; t < 128; ++t)
      m = fmaxf(m, srow[lane + 32 * t]);
#pragma unroll
    for (int off = 16; off > 0; off >>= 1)
      m = fmaxf(m, __shfl_xor(m, off, 32));

    float s = 0.f;
#pragma unroll 4
    for (int t = 0; t < 128; ++t) {
      float e = __expf(srow[lane + 32 * t] - m);
      srow[lane + 32 * t] = e;
      s += e;
    }
#pragma unroll
    for (int off = 16; off > 0; off >>= 1)
      s += __shfl_xor(s, off, 32);
    float inv = 1.0f / s;

    float* grow = ab + (size_t)(i0 + row) * N;
#pragma unroll 4
    for (int t = 0; t < 32; ++t) {
      int col = lane * 4 + 128 * t;
      v4f tmp;
#pragma unroll
      for (int e = 0; e < 4; ++e) tmp[e] = srow[col + e] * inv;
      *(v4f*)(grow + col) = tmp;
    }
  }
}

// ---------------------------------------------------------------------------
// K4: out[b,c,m] = gamma * sum_n x[b,c,n]*att[b,m,n] + x[b,c,m]
// GEMM M=c(256), N=m(4096), K=n(4096) via V_WMMA_F32_16X16X32_BF16.
// Each wave: 1 m-tile x 4 c-tiles (4 accumulators); attention B-operand is
// loaded/converted once per K-step and reused by all 4 WMMAs.
// bf16 16x32 half-slot map = two contiguous 8-float runs per lane
// (k = 8g..8g+7 and 16+8g..16+8g+7) -> 4x b128 loads + cvt per fragment.
// ---------------------------------------------------------------------------
__global__ __launch_bounds__(256) void av_epilogue(
    const float* __restrict__ x, const float* __restrict__ att,
    const float* __restrict__ gamma, float* __restrict__ out)
{
  const int lane = threadIdx.x & 31;
  const int wave = threadIdx.x >> 5;
  const int g = lane >> 4;
  const int r = lane & 15;

  int wg  = blockIdx.x * 8 + wave;   // 0..4095
  int b   = wg >> 10;                // 1024 waves per batch
  int rem = wg & 1023;
  int cs  = rem >> 8;                // 0..3   c-strip of 64
  int mt  = rem & 255;               // 0..255 m-tile
  int c0  = cs * 64;
  int m0  = mt * 16;

  const float* xb = x   + (size_t)b * C * N;
  const float* ar = att + (size_t)b * N * N + (size_t)(m0 + r) * N + 8 * g;

  v8f acc[4] = {};
  for (int n0 = 0; n0 < N; n0 += 32) {
    __builtin_prefetch(ar + n0 + 512, 0, 0);   // stream next att chunk ahead
    v4f b0 = *(const v4f*)(ar + n0);
    v4f b1 = *(const v4f*)(ar + n0 + 4);
    v4f b2 = *(const v4f*)(ar + n0 + 16);
    v4f b3 = *(const v4f*)(ar + n0 + 20);
    v16bf bv;
#pragma unroll
    for (int e = 0; e < 4; ++e) {
      bv[e]      = (__bf16)b0[e];
      bv[4 + e]  = (__bf16)b1[e];
      bv[8 + e]  = (__bf16)b2[e];
      bv[12 + e] = (__bf16)b3[e];
    }
#pragma unroll
    for (int ct = 0; ct < 4; ++ct) {
      const float* xr = xb + (size_t)(c0 + ct * 16 + r) * N + n0 + 8 * g;
      v4f a0 = *(const v4f*)(xr);
      v4f a1 = *(const v4f*)(xr + 4);
      v4f a2 = *(const v4f*)(xr + 16);
      v4f a3 = *(const v4f*)(xr + 20);
      v16bf av;
#pragma unroll
      for (int e = 0; e < 4; ++e) {
        av[e]      = (__bf16)a0[e];
        av[4 + e]  = (__bf16)a1[e];
        av[8 + e]  = (__bf16)a2[e];
        av[12 + e] = (__bf16)a3[e];
      }
      acc[ct] = __builtin_amdgcn_wmma_f32_16x16x32_bf16(
          false, av, false, bv, (short)0, acc[ct], false, false);
    }
  }

  float gm = gamma[0];
#pragma unroll
  for (int ct = 0; ct < 4; ++ct) {
#pragma unroll
    for (int p = 0; p < 8; ++p) {
      int c = c0 + ct * 16 + p + 8 * g;        // C/D layout: vgpr p -> M = p+8g
      size_t idx = (size_t)(b * C + c) * N + (m0 + r);
      out[idx] = fmaf(gm, acc[ct][p], x[idx]);
    }
  }
}

// ---------------------------------------------------------------------------
extern "C" void kernel_launch(void* const* d_in, const int* in_sizes, int n_in,
                              void* d_out, int out_size, void* d_ws, size_t ws_size,
                              hipStream_t stream) {
  const float* x     = (const float*)d_in[0];
  const float* wq    = (const float*)d_in[1];
  const float* bq    = (const float*)d_in[2];
  const float* wk    = (const float*)d_in[3];
  const float* bk    = (const float*)d_in[4];
  const float* gamma = (const float*)d_in[5];

  float* out = (float*)d_out;
  float* att = out + (size_t)Bsz * C * N;          // attention region of d_out

  float* qT = (float*)d_ws;                        // 4 MB, [b][n][c]
  float* kT = qT + (size_t)Bsz * CQ * N;           // 4 MB, [b][n][c]

  proj_qk<<<Bsz * CQ * (N / 256), 256, 0, stream>>>(x, wq, bq, wk, bk, qT, kT);

  size_t smem = (size_t)16 * LDSW * sizeof(float); // 256.25 KB of the 320 KB WGP LDS
  energy_softmax<<<Bsz * 256, 256, smem, stream>>>(qT, kT, att);

  av_epilogue<<<512, 256, 0, stream>>>(x, att, gamma, out);  // 4096 waves
}
